// Temp_Attn_67894843015615
// MI455X (gfx1250) — matrix operand order repaired and checked by reading, not compiled, not run on hardware
//
#include <hip/hip_runtime.h>

// Temporal cross-frame attention for MI455X (gfx1250, wave32, WMMA + TDM).
// Heavy math via V_WMMA_F32_16X16X32_F16 (f32 accum). The two large GEMMs
// (sim = QK^T and ctx = attn*V, ~65 GFLOP combined) stage their shared
// 64x32-half B tiles into LDS with the Tensor Data Mover (double buffered),
// removing the 4x redundant per-wave global B loads seen in round 1.
// B=4, N=3 frames, C=256, Ck=128, HW=2304.

#define HWD 2304
#define CD 256
#define CKD 128
#define NFRAME 12
#define NB 4
#define SIM_SCALE 0.08838834764831845f /* 128^-0.5 */
#define BPITCH 40 /* halves per staged LDS row: 32 data + 8 pad (80B pitch) */

typedef __attribute__((ext_vector_type(16))) _Float16 v16h;
typedef __attribute__((ext_vector_type(8)))  _Float16 v8h;
typedef __attribute__((ext_vector_type(8)))  float    v8f;
typedef __attribute__((ext_vector_type(4)))  unsigned int u32x4;
typedef __attribute__((ext_vector_type(8)))  int i32x8;
typedef __attribute__((ext_vector_type(4)))  int i32x4;

#if __has_builtin(__builtin_amdgcn_tensor_load_to_lds) && \
    __has_builtin(__builtin_amdgcn_s_wait_tensorcnt)
#define HAVE_TDM 1
#else
#define HAVE_TDM 0
#endif

static __device__ __forceinline__ v8f wmma_f16(v16h a, v16h b, v8f c) {
  // D = A(16x32) * B(32x16) + C, f32 accum
  return __builtin_amdgcn_wmma_f32_16x16x32_f16(false, a, false, b, (short)0, c,
                                                false, false);
}

// A fragment (16x32 f16): lane holds row (lane&15); element e -> K =
// (lane<16?0:16)+e.
// p0 = Arow + row*lda + (lane>>4)*16 + kbase ; one contiguous 32B run.
static __device__ __forceinline__ v16h load_a_h(const _Float16* __restrict__ p0) {
  const v8h* p = (const v8h*)p0;
  v8h c0 = p[0];
  v8h c1 = p[1]; // +8 halves
  v16h r;
#pragma unroll
  for (int t = 0; t < 8; ++t) { r[t] = c0[t]; r[8 + t] = c1[t]; }
  return r;
}

// B fragment (32x16 f16) from Bt stored (N,K) row-major: lane holds column
// (lane&15); element e -> K = (lane<16?0:16)+e.
// p0 = Bt + n*ldb + (lane>>4)*16 + kbase ; one contiguous 32B run.
static __device__ __forceinline__ v16h load_b_h(const _Float16* p0) {
  const v8h* p = (const v8h*)p0;
  v8h c0 = p[0];
  v8h c1 = p[1];
  v16h r;
#pragma unroll
  for (int t = 0; t < 8; ++t) { r[t] = c0[t]; r[8 + t] = c1[t]; }
  return r;
}

// A fragment from an f32 row-major matrix (convert on load).
static __device__ __forceinline__ v16h load_a_f(const float* __restrict__ p0) {
  v16h r;
#pragma unroll
  for (int t = 0; t < 8; ++t) {
    r[t] = (_Float16)p0[t];
    r[8 + t] = (_Float16)p0[8 + t];
  }
  return r;
}

#if HAVE_TDM
// Issue one TDM transfer: a 64-row x 32-half 2D tile of Bt (row stride
// ldb halves) -> LDS at byte offset lds_off, rows padded to an 80B pitch
// (pad_enable, pad_interval: 16 dwords, pad_amount: 4 dwords).
// D# layout per CDNA5 ISA 8.3/8.4. Tensor dims set huge so nothing clips.
// This toolchain exposes the 6-arg builtin (g0, g1, g2, g3, i32x8, cpol);
// groups 2/3 and the trailing vector are zero (2-D tile only).
static __device__ __forceinline__ void tdm_stage_b(const _Float16* gsrc,
                                                   unsigned lds_off,
                                                   unsigned ldb) {
  unsigned long long ga = (unsigned long long)(size_t)gsrc;
  u32x4 g0;
  g0[0] = 1u;                                   // count=1, user descriptor
  g0[1] = lds_off;                              // lds_addr (bytes)
  g0[2] = (unsigned)(ga & 0xffffffffu);         // global_addr[31:0]
  g0[3] = (unsigned)((ga >> 32) & 0x01ffffffu)  // global_addr[56:32]
          | (2u << 30);                         // type=2 ("image")
  const unsigned td0 = 1u << 20, td1 = 1u << 20;
  i32x8 g1;
  g1[0] = (int)((1u << 16)    // data_size=1 -> 2 bytes/elem
                | (1u << 20)  // pad_enable
                | (3u << 22)  // pad_interval: 16 dwords (64B) per pad
                | (3u << 25));// pad_amount: 4 dwords (16B)
  g1[1] = (int)((td0 & 0xffffu) << 16);               // tensor_dim0 lo
  g1[2] = (int)((td0 >> 16) | ((td1 & 0xffffu) << 16)); // dim0 hi, dim1 lo
  g1[3] = (int)((td1 >> 16) | (32u << 16));           // dim1 hi, tile_dim0=32
  g1[4] = (int)64u;                                   // tile_dim1=64, dim2=0
  g1[5] = (int)ldb;                                   // tensor_dim0_stride lo
  g1[6] = 0;                                          // stride hi / dim1_stride
  g1[7] = 0;
  i32x4 z4 = {0, 0, 0, 0};
  i32x8 z8 = {0, 0, 0, 0, 0, 0, 0, 0};
  __builtin_amdgcn_tensor_load_to_lds(g0, g1, z4, z4, z8, 0);
}
#endif

// ---------------------------------------------------------------- prep ----
__global__ __launch_bounds__(256) void k_prep(
    const float* __restrict__ Wk, const float* __restrict__ bk,
    const float* __restrict__ gk, const float* __restrict__ betak,
    const float* __restrict__ mk, const float* __restrict__ vk,
    const float* __restrict__ Wv, const float* __restrict__ Wo,
    const float* __restrict__ Wf, _Float16* __restrict__ Wk_h,
    float* __restrict__ kbias, _Float16* __restrict__ Wv_h,
    _Float16* __restrict__ Wo_h, _Float16* __restrict__ WfA_h,
    _Float16* __restrict__ WfB_h) {
  int idx = blockIdx.x * 256 + threadIdx.x;
  if (idx < CKD * CD) { // 32768 : Wk (BN-folded), Wv, Wo (same elem count)
    int ck = idx / CD;
    float inv = gk[ck] * rsqrtf(vk[ck] + 1e-5f);
    Wk_h[idx] = (_Float16)(Wk[idx] * inv);
    Wv_h[idx] = (_Float16)Wv[idx];
    Wo_h[idx] = (_Float16)Wo[idx];
  }
  if (idx < CD * CD) { // 65536 : split Wf (o, 2C) into two (o, C) slabs
    int o = idx / CD, c = idx % CD;
    WfA_h[idx] = (_Float16)Wf[(size_t)o * (2 * CD) + c];
    WfB_h[idx] = (_Float16)Wf[(size_t)o * (2 * CD) + CD + c];
  }
  if (idx < CKD) {
    float inv = gk[idx] * rsqrtf(vk[idx] + 1e-5f);
    kbias[idx] = bk[idx] * inv + betak[idx] - mk[idx] * inv;
  }
}

// ------------------------------------------------- x -> XT (hw, c) f16 ----
__global__ __launch_bounds__(256) void k_transpose(const float* __restrict__ x,
                                                   _Float16* __restrict__ XT) {
  __shared__ float tile[32][33];
  int f = blockIdx.z;
  int hw0 = blockIdx.x * 32, c0 = blockIdx.y * 32;
  int tx = threadIdx.x & 31, ty = threadIdx.x >> 5; // 32 x 8
  const float* src = x + ((size_t)f * CD + c0) * HWD + hw0;
#pragma unroll
  for (int r = 0; r < 32; r += 8)
    tile[ty + r][tx] = src[(size_t)(ty + r) * HWD + tx];
  __syncthreads();
  _Float16* dst = XT + ((size_t)f * HWD + hw0) * CD + c0;
#pragma unroll
  for (int r = 0; r < 32; r += 8)
    dst[(size_t)(ty + r) * CD + tx] = (_Float16)tile[tx][ty + r];
}

// -------------------------------- Kt(hw,ck) = XT(hw,c) x Wk^T + kbias ----
__global__ __launch_bounds__(128) void k_gemm_kt(
    const _Float16* __restrict__ XT, const _Float16* __restrict__ Wk_h,
    const float* __restrict__ kbias, _Float16* __restrict__ Kt) {
  int f = blockIdx.z;
  const _Float16* A = XT + (size_t)f * HWD * CD;
  _Float16* out = Kt + (size_t)f * HWD * CKD;
  int lane = threadIdx.x & 31, wv = threadIdx.x >> 5;
  int m0 = blockIdx.x * 64 + wv * 16, n0 = blockIdx.y * 64;
  int lr = lane & 15, hi = lane >> 4;
  v8f acc[4] = {};
  const _Float16* ap = A + (size_t)(m0 + lr) * CD + hi * 16;
  for (int k = 0; k < CD; k += 32) {
    __builtin_prefetch(ap + k + 64, 0, 3);
    v16h a = load_a_h(ap + k);
#pragma unroll
    for (int t = 0; t < 4; ++t) {
      const _Float16* bp = Wk_h + (size_t)(n0 + t * 16 + lr) * CD + hi * 16 + k;
      acc[t] = wmma_f16(a, load_b_h(bp), acc[t]);
    }
  }
  int rsel = hi * 8;
#pragma unroll
  for (int t = 0; t < 4; ++t) {
    int col = n0 + t * 16 + lr;
    float bias = kbias[col];
#pragma unroll
    for (int g = 0; g < 8; ++g)
      out[(size_t)(m0 + rsel + g) * CKD + col] = (_Float16)(acc[t][g] + bias);
  }
}

// ---------------------------------- Vc(ck,hw) = Wv x X + bv (per-row) ----
__global__ __launch_bounds__(128) void k_gemm_vc(
    const _Float16* __restrict__ Wv_h, const _Float16* __restrict__ XT,
    const float* __restrict__ bv, _Float16* __restrict__ Vc) {
  int f = blockIdx.z;
  const _Float16* Bt = XT + (size_t)f * HWD * CD; // (hw, c) = (N, K)
  _Float16* out = Vc + (size_t)f * CKD * HWD;
  int lane = threadIdx.x & 31, wv = threadIdx.x >> 5;
  int m0 = blockIdx.x * 64 + wv * 16, n0 = blockIdx.y * 64;
  int lr = lane & 15, hi = lane >> 4;
  v8f acc[4] = {};
  const _Float16* ap = Wv_h + (size_t)(m0 + lr) * CD + hi * 16;
  for (int k = 0; k < CD; k += 32) {
    v16h a = load_a_h(ap + k);
#pragma unroll
    for (int t = 0; t < 4; ++t) {
      const _Float16* bp = Bt + (size_t)(n0 + t * 16 + lr) * CD + hi * 16 + k;
      acc[t] = wmma_f16(a, load_b_h(bp), acc[t]);
    }
  }
  int rsel = hi * 8;
#pragma unroll
  for (int t = 0; t < 4; ++t) {
    int col = n0 + t * 16 + lr;
#pragma unroll
    for (int g = 0; g < 8; ++g) {
      int m = m0 + rsel + g;
      out[(size_t)m * HWD + col] = (_Float16)(acc[t][g] + bv[m]);
    }
  }
}

// ------------------- sim(q,k) = Kt_i(q,c) x Kt_j(k,c)^T * scale (f16) ----
// B tile (shared by all 4 waves) staged via TDM into LDS, double buffered.
__global__ __launch_bounds__(128) void k_gemm_sim(
    const _Float16* __restrict__ Kt, _Float16* __restrict__ sim, int fi,
    int fj) {
  int b = blockIdx.z;
  const _Float16* A = Kt + (size_t)(b * 3 + fi) * HWD * CKD;
  const _Float16* Bt = Kt + (size_t)(b * 3 + fj) * HWD * CKD;
  _Float16* out = sim + (size_t)b * HWD * HWD;
  int lane = threadIdx.x & 31, wv = threadIdx.x >> 5;
  int m0 = blockIdx.x * 64 + wv * 16, n0 = blockIdx.y * 64;
  int lr = lane & 15, hi = lane >> 4;
  v8f acc[4] = {};
  const _Float16* ap = A + (size_t)(m0 + lr) * CKD + hi * 16;
#if HAVE_TDM
  __shared__ _Float16 Bs[2][64 * BPITCH];
  // LDS aperture: low 32 bits of the flat address are the LDS byte offset.
  unsigned lds0 = (unsigned)(size_t)(void*)&Bs[0][0];
  unsigned lds1 = (unsigned)(size_t)(void*)&Bs[1][0];
  bool lead = (wv == 0);
  const _Float16* src = Bt + (size_t)n0 * CKD;
  if (lead) tdm_stage_b(src, lds0, CKD);
  __builtin_amdgcn_s_wait_tensorcnt(0);
  __syncthreads();
  for (int ks = 0; ks < CKD / 32; ++ks) {
    int k = ks * 32, cur = ks & 1;
    if (lead && k + 32 < CKD)
      tdm_stage_b(src + k + 32, cur ? lds0 : lds1, CKD);
    __builtin_prefetch(ap + k + 32, 0, 3);
    v16h a = load_a_h(ap + k);
#pragma unroll
    for (int t = 0; t < 4; ++t) {
      const _Float16* bp = &Bs[cur][(t * 16 + lr) * BPITCH + hi * 16];
      acc[t] = wmma_f16(a, load_b_h(bp), acc[t]);
    }
    __builtin_amdgcn_s_wait_tensorcnt(0);
    __syncthreads();
  }
#else
  for (int k = 0; k < CKD; k += 32) {
    __builtin_prefetch(ap + k + 32, 0, 3);
    v16h a = load_a_h(ap + k);
#pragma unroll
    for (int t = 0; t < 4; ++t) {
      const _Float16* bp = Bt + (size_t)(n0 + t * 16 + lr) * CKD + hi * 16 + k;
      acc[t] = wmma_f16(a, load_b_h(bp), acc[t]);
    }
  }
#endif
  int rsel = hi * 8;
#pragma unroll
  for (int t = 0; t < 4; ++t) {
    int col = n0 + t * 16 + lr;
#pragma unroll
    for (int g = 0; g < 8; ++g)
      out[(size_t)(m0 + rsel + g) * HWD + col] =
          (_Float16)(acc[t][g] * SIM_SCALE);
  }
}

// -------- row softmax in place (f16); optionally stream f32 attn maps ----
__global__ __launch_bounds__(256) void k_softmax(_Float16* __restrict__ sim,
                                                 float* __restrict__ maps,
                                                 int write_maps, int jj) {
  int lane = threadIdx.x & 31, wv = threadIdx.x >> 5;
  int row = blockIdx.x * 8 + wv; // 0 .. NB*HWD-1
  int b = row / HWD, q = row % HWD;
  _Float16* r = sim + (size_t)b * HWD * HWD + (size_t)q * HWD;
  float vals[HWD / 32];
  float mx = -3.0e38f;
#pragma unroll
  for (int t = 0; t < HWD / 32; ++t) {
    vals[t] = (float)r[lane + t * 32];
    mx = fmaxf(mx, vals[t]);
  }
#pragma unroll
  for (int off = 16; off > 0; off >>= 1) mx = fmaxf(mx, __shfl_xor(mx, off, 32));
  float s = 0.f;
#pragma unroll
  for (int t = 0; t < HWD / 32; ++t) {
    vals[t] = __expf(vals[t] - mx);
    s += vals[t];
  }
#pragma unroll
  for (int off = 16; off > 0; off >>= 1) s += __shfl_xor(s, off, 32);
  float inv = 1.0f / s;
  float* mp = maps + (((size_t)b * 2 + jj) * HWD + q) * HWD;
#pragma unroll
  for (int t = 0; t < HWD / 32; ++t) {
    float p = vals[t] * inv;
    r[lane + t * 32] = (_Float16)p;
    if (write_maps) mp[lane + t * 32] = p;
  }
}

// ------------- ctx(q,c) (+)= attn(q,k) x Vc_j(c,k)^T, f32 accumulator ----
// Vc tile staged via TDM into LDS (shared by all 4 waves), double buffered.
__global__ __launch_bounds__(128) void k_gemm_ctx(
    const _Float16* __restrict__ attn, const _Float16* __restrict__ Vc,
    float* __restrict__ ctx, int fj, int accum) {
  int b = blockIdx.z;
  const _Float16* A = attn + (size_t)b * HWD * HWD;
  const _Float16* Bt = Vc + (size_t)(b * 3 + fj) * CKD * HWD; // (ck, hw)=(N,K)
  float* out = ctx + (size_t)b * HWD * CKD;
  int lane = threadIdx.x & 31, wv = threadIdx.x >> 5;
  int m0 = blockIdx.x * 64 + wv * 16, n0 = blockIdx.y * 64;
  int lr = lane & 15, hi = lane >> 4;
  v8f acc[4] = {};
  const _Float16* ap = A + (size_t)(m0 + lr) * HWD + hi * 16;
#if HAVE_TDM
  __shared__ _Float16 Bs[2][64 * BPITCH];
  unsigned lds0 = (unsigned)(size_t)(void*)&Bs[0][0];
  unsigned lds1 = (unsigned)(size_t)(void*)&Bs[1][0];
  bool lead = (wv == 0);
  const _Float16* src = Bt + (size_t)n0 * HWD;
  if (lead) tdm_stage_b(src, lds0, HWD);
  __builtin_amdgcn_s_wait_tensorcnt(0);
  __syncthreads();
  for (int ks = 0; ks < HWD / 32; ++ks) {
    int k = ks * 32, cur = ks & 1;
    if (lead && k + 32 < HWD)
      tdm_stage_b(src + k + 32, cur ? lds0 : lds1, HWD);
    __builtin_prefetch(ap + k + 128, 0, 3);
    v16h a = load_a_h(ap + k);
#pragma unroll
    for (int t = 0; t < 4; ++t) {
      const _Float16* bp = &Bs[cur][(t * 16 + lr) * BPITCH + hi * 16];
      acc[t] = wmma_f16(a, load_b_h(bp), acc[t]);
    }
    __builtin_amdgcn_s_wait_tensorcnt(0);
    __syncthreads();
  }
#else
  for (int k = 0; k < HWD; k += 32) {
    __builtin_prefetch(ap + k + 128, 0, 3);
    v16h a = load_a_h(ap + k);
#pragma unroll
    for (int t = 0; t < 4; ++t) {
      const _Float16* bp = Bt + (size_t)(n0 + t * 16 + lr) * HWD + hi * 16 + k;
      acc[t] = wmma_f16(a, load_b_h(bp), acc[t]);
    }
  }
#endif
  int rsel = hi * 8;
#pragma unroll
  for (int t = 0; t < 4; ++t) {
    int col = n0 + t * 16 + lr;
#pragma unroll
    for (int g = 0; g < 8; ++g) {
      size_t idx = (size_t)(m0 + rsel + g) * CKD + col;
      out[idx] = accum ? (out[idx] + acc[t][g]) : acc[t][g];
    }
  }
}

// --------------- matching(q,o) = ctx(q,c) x Wo(o,c)^T + 2*bo   (f16) ----
__global__ __launch_bounds__(128) void k_gemm_match(
    const float* __restrict__ ctx, const _Float16* __restrict__ Wo_h,
    const float* __restrict__ bo, _Float16* __restrict__ match) {
  int b = blockIdx.z;
  const float* A = ctx + (size_t)b * HWD * CKD;
  _Float16* out = match + (size_t)b * HWD * CD;
  int lane = threadIdx.x & 31, wv = threadIdx.x >> 5;
  int m0 = blockIdx.x * 64 + wv * 16, n0 = blockIdx.y * 64;
  int lr = lane & 15, hi = lane >> 4;
  v8f acc[4] = {};
  const float* ap = A + (size_t)(m0 + lr) * CKD + hi * 16;
  for (int k = 0; k < CKD; k += 32) {
    v16h a = load_a_f(ap + k);
#pragma unroll
    for (int t = 0; t < 4; ++t) {
      const _Float16* bp = Wo_h + (size_t)(n0 + t * 16 + lr) * CKD + hi * 16 + k;
      acc[t] = wmma_f16(a, load_b_h(bp), acc[t]);
    }
  }
  int rsel = hi * 8;
#pragma unroll
  for (int t = 0; t < 4; ++t) {
    int col = n0 + t * 16 + lr;
    float bias = 2.0f * bo[col];
#pragma unroll
    for (int g = 0; g < 8; ++g)
      out[(size_t)(m0 + rsel + g) * CD + col] = (_Float16)(acc[t][g] + bias);
  }
}

// out[f,o,q] = x[f,o,q] + XT_i(q,c)xWfA(o,c)^T + match(q,c)xWfB(o,c)^T + bf
__global__ __launch_bounds__(128) void k_gemm_fuse(
    const _Float16* __restrict__ XT, const _Float16* __restrict__ match,
    const _Float16* __restrict__ WfA, const _Float16* __restrict__ WfB,
    const float* __restrict__ bf, const float* __restrict__ x,
    float* __restrict__ out, int fi) {
  int b = blockIdx.z;
  int f = b * 3 + fi;
  const _Float16* A1 = XT + (size_t)f * HWD * CD;
  const _Float16* A2 = match + (size_t)b * HWD * CD;
  int lane = threadIdx.x & 31, wv = threadIdx.x >> 5;
  int m0 = blockIdx.x * 64 + wv * 16, n0 = blockIdx.y * 64;
  int lr = lane & 15, hi = lane >> 4;
  v8f acc[4] = {};
  const _Float16* ap1 = A1 + (size_t)(m0 + lr) * CD + hi * 16;
  for (int k = 0; k < CD; k += 32) {
    v16h a = load_a_h(ap1 + k);
#pragma unroll
    for (int t = 0; t < 4; ++t) {
      const _Float16* bp = WfA + (size_t)(n0 + t * 16 + lr) * CD + hi * 16 + k;
      acc[t] = wmma_f16(a, load_b_h(bp), acc[t]);
    }
  }
  const _Float16* ap2 = A2 + (size_t)(m0 + lr) * CD + hi * 16;
  for (int k = 0; k < CD; k += 32) {
    v16h a = load_a_h(ap2 + k);
#pragma unroll
    for (int t = 0; t < 4; ++t) {
      const _Float16* bp = WfB + (size_t)(n0 + t * 16 + lr) * CD + hi * 16 + k;
      acc[t] = wmma_f16(a, load_b_h(bp), acc[t]);
    }
  }
  int rsel = hi * 8;
#pragma unroll
  for (int t = 0; t < 4; ++t) {
    int col = n0 + t * 16 + lr; // output channel o
    float bias = bf[col];
#pragma unroll
    for (int g = 0; g < 8; ++g) {
      int q = m0 + rsel + g;
      size_t idx = ((size_t)f * CD + col) * HWD + q;
      out[idx] = x[idx] + acc[t][g] + bias;
    }
  }
}

// -------------------------------------------------------------- launch ----
extern "C" void kernel_launch(void* const* d_in, const int* in_sizes, int n_in,
                              void* d_out, int out_size, void* d_ws,
                              size_t ws_size, hipStream_t stream) {
  (void)in_sizes; (void)n_in; (void)out_size; (void)ws_size;
  const float* x = (const float*)d_in[0];
  const float* Wk = (const float*)d_in[1];
  const float* bk = (const float*)d_in[2];
  const float* gk = (const float*)d_in[3];
  const float* betak = (const float*)d_in[4];
  const float* mk = (const float*)d_in[5];
  const float* vk = (const float*)d_in[6];
  const float* Wv = (const float*)d_in[7];
  const float* bv = (const float*)d_in[8];
  const float* Wo = (const float*)d_in[9];
  const float* bo = (const float*)d_in[10];
  const float* Wf = (const float*)d_in[11];
  const float* bf = (const float*)d_in[12];

  float* out = (float*)d_out;                    // (B,3,C,H,W)
  float* maps = out + (size_t)NB * 3 * CD * HWD; // (B,2,HW,HW)

  char* ws = (char*)d_ws;
  size_t off = 0;
  auto wsalloc = [&](size_t bytes) -> void* {
    void* p = ws + off;
    off = (off + bytes + 255) & ~(size_t)255;
    return p;
  };
  _Float16* XT_h   = (_Float16*)wsalloc((size_t)NFRAME * HWD * CD * 2);
  _Float16* Kt     = (_Float16*)wsalloc((size_t)NFRAME * HWD * CKD * 2);
  _Float16* Vc     = (_Float16*)wsalloc((size_t)NFRAME * CKD * HWD * 2);
  _Float16* Wk_h   = (_Float16*)wsalloc((size_t)CKD * CD * 2);
  _Float16* Wv_h   = (_Float16*)wsalloc((size_t)CKD * CD * 2);
  _Float16* Wo_h   = (_Float16*)wsalloc((size_t)CD * CKD * 2);
  _Float16* WfA_h  = (_Float16*)wsalloc((size_t)CD * CD * 2);
  _Float16* WfB_h  = (_Float16*)wsalloc((size_t)CD * CD * 2);
  float*    kbias  = (float*)wsalloc((size_t)CKD * 4);
  _Float16* simb   = (_Float16*)wsalloc((size_t)NB * HWD * HWD * 2);
  float*    ctxb   = (float*)wsalloc((size_t)NB * HWD * CKD * 4);
  _Float16* matchb = (_Float16*)wsalloc((size_t)NB * HWD * CD * 2);

  k_prep<<<256, 256, 0, stream>>>(Wk, bk, gk, betak, mk, vk, Wv, Wo, Wf, Wk_h,
                                  kbias, Wv_h, Wo_h, WfA_h, WfB_h);
  k_transpose<<<dim3(HWD / 32, CD / 32, NFRAME), 256, 0, stream>>>(x, XT_h);
  k_gemm_kt<<<dim3(HWD / 64, CKD / 64, NFRAME), 128, 0, stream>>>(XT_h, Wk_h,
                                                                  kbias, Kt);
  k_gemm_vc<<<dim3(CKD / 64, HWD / 64, NFRAME), 128, 0, stream>>>(Wv_h, XT_h,
                                                                  bv, Vc);
  for (int i = 0; i < 3; ++i) {
    int jj = 0;
    for (int j = 0; j < 3; ++j) {
      if (j == i) continue;
      k_gemm_sim<<<dim3(HWD / 64, HWD / 64, NB), 128, 0, stream>>>(Kt, simb, i,
                                                                   j);
      k_softmax<<<(NB * HWD) / 8, 256, 0, stream>>>(simb, maps,
                                                    (i == 0) ? 1 : 0, jj);
      k_gemm_ctx<<<dim3(HWD / 64, CKD / 64, NB), 128, 0, stream>>>(simb, Vc,
                                                                   ctxb, j, jj);
      ++jj;
    }
    k_gemm_match<<<dim3(HWD / 64, CD / 64, NB), 128, 0, stream>>>(ctxb, Wo_h,
                                                                  bo, matchb);
    k_gemm_fuse<<<dim3(HWD / 64, CD / 64, NB), 128, 0, stream>>>(
        XT_h, matchb, WfA_h, WfB_h, bf, x, out, i);
  }
}
